// Categorical2DSemanticMapModule_54881092108873
// MI455X (gfx1250) — compile-verified, matrix operand order checked by reading, not run on hardware
//
#include <hip/hip_runtime.h>
#include <hip/hip_bf16.h>

// ---------------------------------------------------------------------------
// CDNA5 / gfx1250 implementation of the semantic-map projection module.
//   per frame:  zero voxel grid -> 4x4 mean-pool sem channels -> trilinear
//               atomic scatter-splat -> WMMA masked z-reduction + clip.
// Voxel accumulator (54.4 MB) fits in MI455X's 192 MB L2, so frames are
// processed sequentially reusing one scratch grid; HBM only streams inputs.
// The z-reduction runs on the WMMA pipe (f32 16x16x32 f16 — exact here since
// all A operands are small integers after round()).
// ---------------------------------------------------------------------------

typedef __attribute__((ext_vector_type(16))) _Float16 v16h;
typedef __attribute__((ext_vector_type(8)))  float    v8f;

#define NUM_SEM   16
#define NCH       17                 // 1 occupancy + 16 semantic
#define FRAME_H   480
#define FRAME_W   640
#define H2        120                // FRAME_H / DU_SCALE
#define W2        160                // FRAME_W / DU_SCALE
#define NPTS      (H2 * W2)          // 19200 points per frame
#define VR        100                // VISION_RANGE
#define ZL        80                 // Z_LEVELS
#define VOX_PER_CH (VR * VR * ZL)    // 800000
#define VOX_FLOATS (NCH * VOX_PER_CH)// 13,600,000 floats (54.4 MB)
#define FRAME_CH  20
#define FRAME_FLOATS (FRAME_CH * FRAME_H * FRAME_W)
#define OUT_CH    18
#define OUT_FRAME (OUT_CH * VR * VR) // 180000
#define NTILES    625                // 10000 (x,y) positions / 16 rows per WMMA tile
#define Z_AGENT_LO 8                 // MIN_MAPPED_H
#define Z_AGENT_HI 25                // MAX_MAPPED_H (exclusive)

// ---------------------------------------------------------------------------
// 1) zero the voxel accumulator (float4 stores; stays resident in L2)
// ---------------------------------------------------------------------------
__global__ void zero_vox_kernel(float4* __restrict__ vox) {
    int i = blockIdx.x * blockDim.x + threadIdx.x;
    if (i < VOX_FLOATS / 4) vox[i] = make_float4(0.f, 0.f, 0.f, 0.f);
}

// ---------------------------------------------------------------------------
// 2) 4x4 mean-pool of the 16 semantic channels: [16,480,640] -> [16,120,160]
//    float4 row loads; pure streaming, HBM-bound.
// ---------------------------------------------------------------------------
__global__ void pool_sem_kernel(const float* __restrict__ obsF,
                                float* __restrict__ semp) {
    int idx = blockIdx.x * 256 + threadIdx.x;
    if (idx >= NUM_SEM * NPTS) return;
    int c = idx / NPTS;
    int p = idx - c * NPTS;
    int i = p / W2, j = p - i * W2;
    const float* src = obsF + (size_t)(4 + c) * (FRAME_H * FRAME_W)
                            + (size_t)(4 * i) * FRAME_W + 4 * j;
    float s = 0.f;
#pragma unroll
    for (int a = 0; a < 4; ++a) {
        float4 v = *(const float4*)(src + a * FRAME_W);
        s += v.x + v.y + v.z + v.w;
    }
    semp[idx] = s * (1.0f / 16.0f);
}

// ---------------------------------------------------------------------------
// 3) trilinear scatter-splat. One thread per camera point; 8 corners x 17
//    channels of global_atomic_add_f32 (resolved in L2 — grid is L2-resident).
// ---------------------------------------------------------------------------
__global__ void splat_kernel(const float* __restrict__ obsF,
                             const float* __restrict__ semp,
                             float* __restrict__ vox) {
    int p = blockIdx.x * 256 + threadIdx.x;
    if (p >= NPTS) return;

    // gfx1250 path: warms L2 for the feature gather below (global_prefetch_b8)
    __builtin_prefetch(semp + p, 0, 0);

    int i = p / W2, j = p - i * W2;
    float d = obsF[3 * (FRAME_H * FRAME_W) + (4 * i) * FRAME_W + 4 * j];
    // invalid depth -> reference coords=2.0 -> all trilinear weights 0 -> skip
    if (!(d > 20.0f && d < 500.0f)) return;

    const float F_PIX = 320.0f / tanf(0.6894250515f); // 640/2 / tan(39.5 deg)
    float invF = 1.0f / F_PIX;
    float X = (4.0f * (float)j - 319.5f) * d * invF;
    float Z = ((479.0f - 4.0f * (float)i) - 239.5f) * d * invF;

    // pos in voxel units (the +-dims/2 normalization cancels exactly)
    float posd[3];
    posd[0] = (X + 250.0f) * 0.2f;          // x
    posd[1] = d * 0.2f;                     // depth/y
    posd[2] = (Z + 88.0f) * 0.2f + 8.0f;    // z (height)
    const int dims[3] = { VR, VR, ZL };

    float wts[3][2];
    int   ix[3][2];
#pragma unroll
    for (int dd = 0; dd < 3; ++dd) {
        float b = floorf(posd[dd]);
#pragma unroll
        for (int t = 0; t < 2; ++t) {
            float pp = b + (float)t;
            bool safe = (pp > 0.0f) && (pp < (float)dims[dd]); // strict, per ref
            wts[dd][t] = safe ? (1.0f - fabsf(posd[dd] - pp)) : 0.0f;
            ix[dd][t]  = safe ? (int)pp : 0;
        }
    }

    float feat[NCH];
    feat[0] = 1.0f;
#pragma unroll
    for (int c = 0; c < NUM_SEM; ++c) feat[1 + c] = semp[c * NPTS + p];

#pragma unroll
    for (int i0 = 0; i0 < 2; ++i0)
#pragma unroll
    for (int i1 = 0; i1 < 2; ++i1)
#pragma unroll
    for (int i2 = 0; i2 < 2; ++i2) {
        float w = wts[0][i0] * wts[1][i1] * wts[2][i2];
        if (w == 0.0f) continue;
        int idx = (ix[0][i0] * VR + ix[1][i1]) * ZL + ix[2][i2];
#pragma unroll
        for (int c = 0; c < NCH; ++c)
            atomicAdd(vox + (size_t)c * VOX_PER_CH + idx, feat[c] * w);
    }
}

// ---------------------------------------------------------------------------
// 4) masked z-reduction on the WMMA pipe.
//    D[16 pos x 16] += round(V)[16 pos x 32 z] * Mask[32 z x 16], 3 K-chunks.
//    N=0 column: all-z mask (exploration); N=1 column: agent-height mask.
//    f16 A operands are exact (small integers after round); f32 accumulate.
//    One wave per (tile, channel); EXEC all-ones at every WMMA.
// ---------------------------------------------------------------------------
__global__ void reduce_wmma_kernel(const float* __restrict__ vox,
                                   float* __restrict__ outF) {
    int lane = threadIdx.x & 31;
    int wave = threadIdx.x >> 5;
    int tile = blockIdx.x * 8 + wave;      // wave-uniform predicate
    if (tile >= NTILES) return;
    int c = blockIdx.y;                    // 0..16

    const float* vbase = vox + (size_t)c * VOX_PER_CH + (size_t)tile * 16 * ZL;
    int m     = lane & 15;                 // A row (position within tile)
    int n     = lane & 15;                 // B column
    int half8 = (lane >> 4) * 8;           // K-range split between lane halves
    const float* row = vbase + m * ZL;

    v8f acc = {};
#pragma unroll
    for (int t = 0; t < 3; ++t) {
        int zb = 32 * t + half8;
        v16h a;
        // A: ISA 16-bit 16x32 layout — element e holds K = zb + e (+8 for e>=8)
#pragma unroll
        for (int run = 0; run < 2; ++run) {
            int z0 = zb + run * 16;        // wave-uniform in/out-of-range
            if (z0 < ZL) {
                float4 x = *(const float4*)(row + z0);
                float4 y = *(const float4*)(row + z0 + 4);
                a[run * 8 + 0] = (_Float16)__builtin_rintf(x.x);
                a[run * 8 + 1] = (_Float16)__builtin_rintf(x.y);
                a[run * 8 + 2] = (_Float16)__builtin_rintf(x.z);
                a[run * 8 + 3] = (_Float16)__builtin_rintf(x.w);
                a[run * 8 + 4] = (_Float16)__builtin_rintf(y.x);
                a[run * 8 + 5] = (_Float16)__builtin_rintf(y.y);
                a[run * 8 + 6] = (_Float16)__builtin_rintf(y.z);
                a[run * 8 + 7] = (_Float16)__builtin_rintf(y.w);
            } else {
#pragma unroll
                for (int e = 0; e < 8; ++e) a[run * 8 + e] = (_Float16)0.0f;
            }
        }
        // B: same K(e, lane-half) mapping; columns 0/1 carry the two masks
        v16h b;
#pragma unroll
        for (int e = 0; e < 16; ++e) {
            int z = zb + (e < 8 ? e : 8 + e);
            float mv = 0.0f;
            if (n == 0)      mv = (z < ZL) ? 1.0f : 0.0f;
            else if (n == 1) mv = (z >= Z_AGENT_LO && z < Z_AGENT_HI) ? 1.0f : 0.0f;
            b[e] = (_Float16)mv;
        }
        acc = __builtin_amdgcn_wmma_f32_16x16x32_f16(
                  false, a, false, b, (short)0, acc, false, false);
    }

    // D layout: VGPR r -> M = r + 8*(lane>>4), N = lane&15.
    if (n <= 1) {
        int mbase = tile * 16 + half8;
        float scale = (c == 0) ? 1.0f : 0.2f;   // MAP/EXP thr = 1, CAT thr = 5
#pragma unroll
        for (int r = 0; r < 8; ++r) {
            float cv = fminf(fmaxf(acc[r] * scale, 0.0f), 1.0f);
            int pos = mbase + r;
            if (c == 0) {
                if (n == 1) outF[pos] = cv;               // fp_map (agent z)
                else        outF[VR * VR + pos] = cv;     // fp_exp (all z)
            } else if (n == 1) {
                outF[(1 + c) * VR * VR + pos] = cv;       // semantic channels
            }
        }
    }
}

// ---------------------------------------------------------------------------
// Host: sequential per-frame pipeline on `stream`, reusing one L2-resident
// scratch grid.  ws layout: [voxels: 13.6M floats][sem_pooled: 307200 floats]
// (~53 MB total).
// ---------------------------------------------------------------------------
extern "C" void kernel_launch(void* const* d_in, const int* in_sizes, int n_in,
                              void* d_out, int out_size, void* d_ws, size_t ws_size,
                              hipStream_t stream) {
    (void)n_in; (void)out_size; (void)ws_size;
    const float* obs = (const float*)d_in[0];
    float* out  = (float*)d_out;
    float* vox  = (float*)d_ws;
    float* semp = vox + VOX_FLOATS;

    int nframes = in_sizes[0] / FRAME_FLOATS;   // B*T = 16

    dim3 blk(256);
    dim3 gZero((VOX_FLOATS / 4 + 255) / 256);
    dim3 gPool((NUM_SEM * NPTS + 255) / 256);
    dim3 gSplat((NPTS + 255) / 256);
    dim3 gRed((NTILES + 7) / 8, NCH);

    for (int f = 0; f < nframes; ++f) {
        const float* obsF = obs + (size_t)f * FRAME_FLOATS;
        float* outF = out + (size_t)f * OUT_FRAME;
        zero_vox_kernel<<<gZero, blk, 0, stream>>>((float4*)vox);
        pool_sem_kernel<<<gPool, blk, 0, stream>>>(obsF, semp);
        splat_kernel<<<gSplat, blk, 0, stream>>>(obsF, semp, vox);
        reduce_wmma_kernel<<<gRed, blk, 0, stream>>>(vox, outF);
    }
}
